// GNNEncoder_36928128811714
// MI455X (gfx1250) — matrix operand order repaired and checked by reading, not compiled, not run on hardware
//
#include <hip/hip_runtime.h>

// ---------------------------------------------------------------------------
// Types for CDNA5 WMMA (wave32)
// ---------------------------------------------------------------------------
typedef __attribute__((ext_vector_type(16))) __bf16 v16bf;
typedef __attribute__((ext_vector_type(8)))  __bf16 v8bf;
typedef __attribute__((ext_vector_type(4)))  __bf16 v4bf;
typedef __attribute__((ext_vector_type(8)))  float  v8f;

union F16frag { v16bf v; v8bf h[2]; };

// Split fp32 into bf16 hi (truncate) + bf16 lo (RNE of residual).
__device__ __forceinline__ void split_bf16(float x, __bf16& hi, __bf16& lo) {
  unsigned u = __builtin_bit_cast(unsigned, x);
  unsigned short hs = (unsigned short)(u >> 16);
  float hf = __builtin_bit_cast(float, u & 0xFFFF0000u);
  float r  = x - hf;
  unsigned v = __builtin_bit_cast(unsigned, r);
  unsigned short ls = (unsigned short)((v + 0x7FFFu + ((v >> 16) & 1u)) >> 16);
  hi = __builtin_bit_cast(__bf16, hs);
  lo = __builtin_bit_cast(__bf16, ls);
}

// ---------------------------------------------------------------------------
// Split-precision bf16 WMMA GEMM:  C[slot] = A(MxK) @ B_slot(KxBN)
// slot = blockIdx.y in [0,5): 0..3 -> Wrel[slot], 4 -> Wroot.
// BM=128 (8 waves x 16 rows), BK=32, BN = 128 (layer1) or 32 (layer2).
// Double-buffered LDS; next tile's global loads overlap WMMA compute.
// ---------------------------------------------------------------------------
template<int BN>
__launch_bounds__(256)
__global__ void gemm_split_kernel(const float* __restrict__ A,
                                  const float* __restrict__ Wrel,
                                  const float* __restrict__ Wroot,
                                  float* __restrict__ Cbase,
                                  int M, int K) {
  constexpr int NT  = BN / 16;
  constexpr int NB  = BN / 32;         // float4's of B per thread per tile
  constexpr int LDT = 40;              // 32 K-elems + 8 pad (bf16) = 80B stride

  __shared__ __align__(16) __bf16 Ah[2][128 * LDT];
  __shared__ __align__(16) __bf16 Al[2][128 * LDT];
  __shared__ __align__(16) __bf16 Bh[2][BN * LDT];
  __shared__ __align__(16) __bf16 Bl[2][BN * LDT];

  const int slot = blockIdx.y;
  const float* __restrict__ B = (slot < 4) ? (Wrel + (size_t)slot * K * BN) : Wroot;
  float* __restrict__ C = Cbase + (size_t)slot * (size_t)M * BN;

  const int row0 = blockIdx.x * 128;
  const int t    = threadIdx.x;
  const int lane = t & 31;
  const int wv   = t >> 5;
  const int lr   = lane & 15;
  const int kh   = lane >> 4;

  // staging registers for the next k-tile (prefetched ahead of compute)
  float4 areg[4];
  float4 breg[NB];

  auto load_tile = [&](int k0) {
    const int kc = (t & 7) * 4;
#pragma unroll
    for (int i = 0; i < 4; ++i) {
      const int gr = row0 + (t >> 3) + i * 32;
      areg[i] = make_float4(0.f, 0.f, 0.f, 0.f);
      if (gr < M) areg[i] = *(const float4*)(A + (size_t)gr * K + k0 + kc);
    }
    const int kr = t >> 3;
#pragma unroll
    for (int j = 0; j < NB; ++j)
      breg[j] = *(const float4*)(B + (size_t)(k0 + kr) * BN + (t & 7) * 4 + j * 32);
  };

  auto store_tile = [&](int buf) {
    const int kc = (t & 7) * 4;
#pragma unroll
    for (int i = 0; i < 4; ++i) {
      const int row = (t >> 3) + i * 32;
      v4bf h4, l4;
      __bf16 h, l;
      split_bf16(areg[i].x, h, l); h4[0] = h; l4[0] = l;
      split_bf16(areg[i].y, h, l); h4[1] = h; l4[1] = l;
      split_bf16(areg[i].z, h, l); h4[2] = h; l4[2] = l;
      split_bf16(areg[i].w, h, l); h4[3] = h; l4[3] = l;
      *(v4bf*)(&Ah[buf][row * LDT + kc]) = h4;
      *(v4bf*)(&Al[buf][row * LDT + kc]) = l4;
    }
    const int kr = t >> 3;
#pragma unroll
    for (int j = 0; j < NB; ++j) {
      const int c0 = (t & 7) * 4 + j * 32;
      __bf16 h, l;
      split_bf16(breg[j].x, h, l); Bh[buf][(c0 + 0) * LDT + kr] = h; Bl[buf][(c0 + 0) * LDT + kr] = l;
      split_bf16(breg[j].y, h, l); Bh[buf][(c0 + 1) * LDT + kr] = h; Bl[buf][(c0 + 1) * LDT + kr] = l;
      split_bf16(breg[j].z, h, l); Bh[buf][(c0 + 2) * LDT + kr] = h; Bl[buf][(c0 + 2) * LDT + kr] = l;
      split_bf16(breg[j].w, h, l); Bh[buf][(c0 + 3) * LDT + kr] = h; Bl[buf][(c0 + 3) * LDT + kr] = l;
    }
  };

  v8f zero;
#pragma unroll
  for (int i = 0; i < 8; ++i) zero[i] = 0.0f;
  v8f acc[NT];
#pragma unroll
  for (int q = 0; q < NT; ++q) acc[q] = zero;

  // prologue: stage tile 0 into LDS buffer 0
  load_tile(0);
  store_tile(0);
  int cur = 0;

  for (int k0 = 0; k0 < K; k0 += 32) {
    __syncthreads();                       // LDS[cur] ready for all waves
    const bool more = (k0 + 32) < K;
    if (more) load_tile(k0 + 32);          // global loads in flight during WMMA

    // ---- A fragments (ISA 16-bit A 16x32 layout) ----
    F16frag ah, al;
    const int abase = (wv * 16 + lr) * LDT;
    ah.h[0] = *(const v8bf*)(&Ah[cur][abase + kh * 8]);
    ah.h[1] = *(const v8bf*)(&Ah[cur][abase + 16 + kh * 8]);
    al.h[0] = *(const v8bf*)(&Al[cur][abase + kh * 8]);
    al.h[1] = *(const v8bf*)(&Al[cur][abase + 16 + kh * 8]);

#pragma unroll
    for (int nt = 0; nt < NT; ++nt) {
      // B fragment: lanes 0-15 hold K=0..7,16..23, lanes 16-31 hold K=8..15,24..31
      F16frag bh, bl;
      const int bbase = (nt * 16 + lr) * LDT + kh * 8;
      bh.h[0] = *(const v8bf*)(&Bh[cur][bbase]);
      bh.h[1] = *(const v8bf*)(&Bh[cur][bbase + 16]);
      bl.h[0] = *(const v8bf*)(&Bl[cur][bbase]);
      bl.h[1] = *(const v8bf*)(&Bl[cur][bbase + 16]);
      // split-precision: hi*hi + hi*lo + lo*hi, fp32 accumulate
      acc[nt] = __builtin_amdgcn_wmma_f32_16x16x32_bf16(false, ah.v, false, bh.v,
                                                        (short)0, acc[nt], false, false);
      acc[nt] = __builtin_amdgcn_wmma_f32_16x16x32_bf16(false, ah.v, false, bl.v,
                                                        (short)0, acc[nt], false, false);
      acc[nt] = __builtin_amdgcn_wmma_f32_16x16x32_bf16(false, al.v, false, bh.v,
                                                        (short)0, acc[nt], false, false);
    }

    if (more) store_tile(cur ^ 1);         // convert + stage next tile
    cur ^= 1;
  }

  // ---- epilogue: C layout VGPR i -> M = kh*8 + i, N = lane&15 ----
#pragma unroll
  for (int nt = 0; nt < NT; ++nt) {
#pragma unroll
    for (int i = 0; i < 8; ++i) {
      const int row = row0 + wv * 16 + kh * 8 + i;
      const int col = nt * 16 + lr;
      if (row < M) C[(size_t)row * BN + col] = acc[nt][i];
    }
  }
}

// ---------------------------------------------------------------------------
// Graph preprocessing: counts, norms, CSR build
// ---------------------------------------------------------------------------
__global__ void zero_i32(int* p, int n) {
  int i = blockIdx.x * blockDim.x + threadIdx.x;
  if (i < n) p[i] = 0;
}

__global__ void count_kernel(const int* __restrict__ dst, const int* __restrict__ et,
                             int* __restrict__ cnt, int E) {
  int e = blockIdx.x * blockDim.x + threadIdx.x;
  if (e < E) atomicAdd(&cnt[dst[e] * 4 + et[e]], 1);
}

__global__ void norm_kernel(const int* __restrict__ cnt, float* __restrict__ nrm,
                            int* __restrict__ rowcnt, int n) {
  int d = blockIdx.x * blockDim.x + threadIdx.x;
  if (d < n) {
    int s = 0;
#pragma unroll
    for (int r = 0; r < 4; ++r) {
      int c = cnt[d * 4 + r];
      s += c;
      nrm[d * 4 + r] = 1.0f / (float)(c > 0 ? c : 1);
    }
    rowcnt[d] = s;
  }
}

__launch_bounds__(1024)
__global__ void scan_kernel(const int* __restrict__ rowcnt, int* __restrict__ rowptr,
                            int* __restrict__ cursor, int n) {
  __shared__ int part[1024];
  const int t = threadIdx.x;
  const int chunk = (n + 1023) >> 10;
  const int start = t * chunk;
  const int end = min(n, start + chunk);
  int s = 0;
  for (int i = start; i < end; ++i) s += rowcnt[i];
  part[t] = s;
  __syncthreads();
  for (int off = 1; off < 1024; off <<= 1) {
    int v = (t >= off) ? part[t - off] : 0;
    __syncthreads();
    part[t] += v;
    __syncthreads();
  }
  int base = (t == 0) ? 0 : part[t - 1];
  for (int i = start; i < end; ++i) {
    rowptr[i] = base;
    cursor[i] = base;
    base += rowcnt[i];
  }
  if (t == 0) rowptr[n] = part[1023];
}

__global__ void fill_kernel(const int* __restrict__ src, const int* __restrict__ dst,
                            const int* __restrict__ et, int* __restrict__ cursor,
                            int* __restrict__ epack, int E) {
  int e = blockIdx.x * blockDim.x + threadIdx.x;
  if (e < E) {
    int d = dst[e];
    int pos = atomicAdd(&cursor[d], 1);
    epack[pos] = src[e] | (et[e] << 24);   // src < 2^24, rel < 4
  }
}

// ---------------------------------------------------------------------------
// Gather-based aggregation (no float atomics): one wave32 per dst node.
// Layer 1: 128 cols (float4/lane), + root + bias, ReLU.
// ---------------------------------------------------------------------------
__launch_bounds__(256)
__global__ void agg_relu128_kernel(const float* __restrict__ H, const int* __restrict__ rowptr,
                                   const int* __restrict__ epack, const float* __restrict__ nrm,
                                   const float* __restrict__ bias, float* __restrict__ out, int n) {
  const int lane = threadIdx.x & 31;
  const int d = blockIdx.x * 8 + (threadIdx.x >> 5);
  if (d >= n) return;
  const float4 b  = *(const float4*)(bias + lane * 4);
  const float4 xr = *(const float4*)(H + ((size_t)4 * n + d) * 128 + lane * 4);
  float ax = xr.x + b.x, ay = xr.y + b.y, az = xr.z + b.z, aw = xr.w + b.w;
  const int e0 = rowptr[d], e1 = rowptr[d + 1];
  for (int e = e0; e < e1; ++e) {
    __builtin_prefetch(epack + e + 8, 0, 0);
    const int p = epack[e];
    const int s = p & 0xFFFFFF;
    const int r = p >> 24;
    const float w = nrm[d * 4 + r];
    const float4 v = *(const float4*)(H + ((size_t)r * n + s) * 128 + lane * 4);
    ax = fmaf(w, v.x, ax); ay = fmaf(w, v.y, ay);
    az = fmaf(w, v.z, az); aw = fmaf(w, v.w, aw);
  }
  float4 o;
  o.x = fmaxf(ax, 0.f); o.y = fmaxf(ay, 0.f);
  o.z = fmaxf(az, 0.f); o.w = fmaxf(aw, 0.f);
  *(float4*)(out + (size_t)d * 128 + lane * 4) = o;
}

// Layer 2 (final): 32 cols (1 float/lane), + root + bias, no ReLU.
__launch_bounds__(256)
__global__ void agg_out32_kernel(const float* __restrict__ H, const int* __restrict__ rowptr,
                                 const int* __restrict__ epack, const float* __restrict__ nrm,
                                 const float* __restrict__ bias, float* __restrict__ out, int n) {
  const int lane = threadIdx.x & 31;
  const int d = blockIdx.x * 8 + (threadIdx.x >> 5);
  if (d >= n) return;
  float a = H[((size_t)4 * n + d) * 32 + lane] + bias[lane];
  const int e0 = rowptr[d], e1 = rowptr[d + 1];
  for (int e = e0; e < e1; ++e) {
    __builtin_prefetch(epack + e + 8, 0, 0);
    const int p = epack[e];
    const int s = p & 0xFFFFFF;
    const int r = p >> 24;
    a = fmaf(nrm[d * 4 + r], H[((size_t)r * n + s) * 32 + lane], a);
  }
  out[(size_t)d * 32 + lane] = a;
}

// ---------------------------------------------------------------------------
// Launcher
// ---------------------------------------------------------------------------
extern "C" void kernel_launch(void* const* d_in, const int* in_sizes, int n_in,
                              void* d_out, int out_size, void* d_ws, size_t ws_size,
                              hipStream_t stream) {
  (void)n_in; (void)out_size; (void)ws_size;
  const float* x     = (const float*)d_in[0];
  const int*   eidx  = (const int*)d_in[1];
  const int*   etype = (const int*)d_in[2];
  const float* W1    = (const float*)d_in[3];
  const float* root1 = (const float*)d_in[4];
  const float* b1    = (const float*)d_in[5];
  const float* W2    = (const float*)d_in[6];
  const float* root2 = (const float*)d_in[7];
  const float* b2    = (const float*)d_in[8];

  const int IN = 1024, HID = 128, OUT = 32, R = 4;
  const int N = in_sizes[0] / IN;
  const int E = in_sizes[2];
  const int* src = eidx;
  const int* dst = eidx + E;

  // workspace carve-out (256B-aligned)
  char* w = (char*)d_ws;
  size_t off = 0;
  auto take = [&](size_t bytes) -> char* {
    char* p = w + off;
    off = (off + bytes + 255) & ~(size_t)255;
    return p;
  };
  int*   cnt    = (int*)  take((size_t)N * R * 4);
  float* nrm    = (float*)take((size_t)N * R * 4);
  int*   rowcnt = (int*)  take((size_t)N * 4);
  int*   rowptr = (int*)  take(((size_t)N + 1) * 4);
  int*   cursor = (int*)  take((size_t)N * 4);
  int*   epack  = (int*)  take((size_t)E * 4);
  float* H1     = (float*)take((size_t)5 * N * HID * 4);   // slots 0..3: W1[r], 4: root1
  float* hbuf   = (float*)take((size_t)N * HID * 4);
  float* H2     = (float*)take((size_t)5 * N * OUT * 4);

  // --- CSR / normalization prep (shared by both layers) ---
  zero_i32<<<(N * R + 255) / 256, 256, 0, stream>>>(cnt, N * R);
  count_kernel<<<(E + 255) / 256, 256, 0, stream>>>(dst, etype, cnt, E);
  norm_kernel<<<(N + 255) / 256, 256, 0, stream>>>(cnt, nrm, rowcnt, N);
  scan_kernel<<<1, 1024, 0, stream>>>(rowcnt, rowptr, cursor, N);
  fill_kernel<<<(E + 255) / 256, 256, 0, stream>>>(src, dst, etype, cursor, epack, E);

  // --- layer 1: projections (WMMA) + gather-aggregate + root + bias + ReLU ---
  dim3 g1((N + 127) / 128, 5, 1);
  gemm_split_kernel<128><<<g1, 256, 0, stream>>>(x, W1, root1, H1, N, IN);
  agg_relu128_kernel<<<(N + 7) / 8, 256, 0, stream>>>(H1, rowptr, epack, nrm, b1, hbuf, N);

  // --- layer 2: projections (WMMA) + gather-aggregate + root + bias ---
  gemm_split_kernel<32><<<g1, 256, 0, stream>>>(hbuf, W2, root2, H2, N, HID);
  agg_out32_kernel<<<(N + 7) / 8, 256, 0, stream>>>(H2, rowptr, epack, nrm, b2,
                                                    (float*)d_out, N);
}
